// EnhancedTFT_28458453303925
// MI455X (gfx1250) — matrix operand order repaired and checked by reading, not compiled, not run on hardware
//
#include <hip/hip_runtime.h>

// ---------------------------------------------------------------------------
// CDNA5 (gfx1250) WMMA implementation of the Enhanced-TFT forward pass.
// bf16 A/B fragments, f32 accumulation via v_wmma_f32_16x16x32_bf16.
// All weights are pre-converted once per launch to bf16, N-major (K-contig),
// so every GEMM inner loop is pure contiguous bf16 loads + WMMA.
// ---------------------------------------------------------------------------

typedef __attribute__((ext_vector_type(16))) __bf16        v16bf;
typedef __attribute__((ext_vector_type(16))) unsigned short v16us;
typedef __attribute__((ext_vector_type(8)))  float          v8f;

#define DEV static __device__ __forceinline__

DEV unsigned short f2bf(float f) {                // RNE float -> bf16
  unsigned u = __builtin_bit_cast(unsigned, f);
  unsigned r = u + 0x7FFFu + ((u >> 16) & 1u);
  return (unsigned short)(r >> 16);
}
DEV float bf2f(unsigned short h) {
  unsigned u = ((unsigned)h) << 16;
  return __builtin_bit_cast(float, u);
}
DEV float sigmoidf_(float x) { return 1.0f / (1.0f + __expf(-x)); }
DEV float eluf_(float x) { return x > 0.f ? x : (__expf(x) - 1.f); }

DEV v8f vzero8() {
  v8f z;
#pragma unroll
  for (int i = 0; i < 8; ++i) z[i] = 0.f;
  return z;
}

// A fragment: 16x32 bf16 tile, row-major source.
// Lane m<16 holds 16 contiguous K values starting at 16*half.
DEV void ldA(v16us& e, const unsigned short* base, int lda) {
  int lane = threadIdx.x & 31;
  const unsigned short* p = base + (long)(lane & 15) * lda + ((lane >> 4) ? 16 : 0);
#pragma unroll
  for (int v = 0; v < 8; ++v) {
    int kb = 2 * v;
    e[2 * v]     = p[kb];
    e[2 * v + 1] = p[kb + 1];
  }
}
// B fragment: weight pre-converted to bf16, N-major (row n = K contiguous).
// Lane n<16 reads 16 contiguous K values starting at k0 + 16*half.
DEV void ldB_bf(v16us& e, const unsigned short* base, int ldk) {
  int lane = threadIdx.x & 31;
  const unsigned short* p = base + (long)(lane & 15) * ldk + ((lane >> 4) ? 16 : 0);
#pragma unroll
  for (int j = 0; j < 16; ++j) e[j] = p[j];
}
DEV v8f wmma_bf16(const v16us& a, const v16us& b, v8f c) {
  return __builtin_amdgcn_wmma_f32_16x16x32_bf16(
      false, __builtin_bit_cast(v16bf, a),
      false, __builtin_bit_cast(v16bf, b),
      (short)0, c, false, false);
}

// ---------------------------------------------------------------------------
// Weight conversion kernels (run once per launch; deterministic).
// ---------------------------------------------------------------------------
__global__ __launch_bounds__(256) void k_conv_copy(const float* src,
                                                   unsigned short* dst, int n) {
  int i = blockIdx.x * 256 + threadIdx.x;
  if (i < n) dst[i] = f2bf(src[i]);
}
// src is (K,N) row-major; dst is (N,K) row-major (coalesced writes).
__global__ __launch_bounds__(256) void k_conv_tr(const float* src,
                                                 unsigned short* dst, int K,
                                                 int N) {
  long i = (long)blockIdx.x * 256 + threadIdx.x;
  long base = (long)blockIdx.y * K * N;      // batch (e.g. per-feature)
  if (i < (long)K * N) {
    int nn = (int)(i / K), kk = (int)(i % K);
    dst[base + i] = f2bf(src[base + (long)kk * N + nn]);
  }
}

// ---------------------------------------------------------------------------
// Generic WMMA GEMM:  out = act(A@Bw^T-layout + bias + bias2 + resid)
// A: M x K bf16 (lda).  Bw: bf16, N-major (N rows of K contiguous).
// Grid: (M/32, N/64); block 256 threads = 8 waves, each wave one 16x16 tile.
// M % 32 == 0, N % 64 == 0, K % 32 == 0 (all shapes here comply).
// ---------------------------------------------------------------------------
__global__ __launch_bounds__(256) void k_gemm(
    const unsigned short* A, int lda, const unsigned short* Bw,
    const float* bias, const float* bias2, const float* resid, int ldr,
    int act, float* outF, unsigned short* outBF, int ldo, int K) {
  int wave = threadIdx.x >> 5;
  long mt = (long)blockIdx.x * 32 + (wave >> 2) * 16;
  long nt = (long)blockIdx.y * 64 + (wave & 3) * 16;
  v8f acc = vzero8();
  for (int k0 = 0; k0 < K; k0 += 32) {
    v16us af, bfr;
    ldA(af, A + mt * lda + k0, lda);
    ldB_bf(bfr, Bw + nt * (long)K + k0, K);
    acc = wmma_bf16(af, bfr, acc);
  }
  int lane = threadIdx.x & 31, half = lane >> 4;
  long n = nt + (lane & 15);
  float badd = (bias ? bias[n] : 0.f) + (bias2 ? bias2[n] : 0.f);
#pragma unroll
  for (int r = 0; r < 8; ++r) {
    long m = mt + r + 8 * half;
    float v = acc[r] + badd;
    if (resid) v += resid[m * ldr + n];
    if (act == 1) v = eluf_(v);
    if (outF)  outF[m * (long)ldo + n] = v;
    if (outBF) outBF[m * (long)ldo + n] = f2bf(v);
  }
}

// ---------------------------------------------------------------------------
// VSN stage 1: h1 = elu(x[...,None]*fc1_w + fc1_b), bf16, [BT, F, H]
// ---------------------------------------------------------------------------
__global__ __launch_bounds__(256) void k_vsn_h1(const float* x, const float* w1,
                                                const float* b1,
                                                unsigned short* h1) {
  long i = (long)blockIdx.x * 256 + threadIdx.x;  // BT*F*H total
  int hc = (int)(i & 255);
  long r = i >> 8;                                // bt*F + f
  int f = (int)(r % 23);
  long bt = r / 23;
  float z = x[bt * 23 + f] * w1[f * 256 + hc] + b1[f * 256 + hc];
  h1[i] = f2bf(eluf_(z));
}

// ---------------------------------------------------------------------------
// VSN fused: per (16-row, feature) block: dual WMMA GEMM (fc2,gate) + gate +
// skip + LayerNorm(H) + feature-softmax scale -> proc bf16 [BT, F*H].
// Grid: (BT/16, F); block 256. w2/wg: bf16 N-major per-feature (f*65536).
// ---------------------------------------------------------------------------
__global__ __launch_bounds__(256) void k_vsn_fused(
    const unsigned short* h1, const float* x, const unsigned short* w2,
    const float* b2, const unsigned short* wg, const float* bg,
    const float* wsk, const float* bsk, const float* lng, const float* lnb,
    const float* fw, unsigned short* proc) {
  __shared__ float s[16][256];
  const int f = blockIdx.y;
  const long bt0 = (long)blockIdx.x * 16;
  int wave = threadIdx.x >> 5, lane = threadIdx.x & 31;
  int half = lane >> 4, col = lane & 15, n0 = wave * 32;
  v8f aF[2] = {vzero8(), vzero8()}, aG[2] = {vzero8(), vzero8()};
  const unsigned short* Ab = h1 + bt0 * 5888 + f * 256;
  const unsigned short* W2b = w2 + (long)f * 65536;
  const unsigned short* WGb = wg + (long)f * 65536;
  for (int k0 = 0; k0 < 256; k0 += 32) {
    v16us af, b;
    ldA(af, Ab + k0, 5888);
    ldB_bf(b, W2b + (long)(n0)      * 256 + k0, 256); aF[0] = wmma_bf16(af, b, aF[0]);
    ldB_bf(b, W2b + (long)(n0 + 16) * 256 + k0, 256); aF[1] = wmma_bf16(af, b, aF[1]);
    ldB_bf(b, WGb + (long)(n0)      * 256 + k0, 256); aG[0] = wmma_bf16(af, b, aG[0]);
    ldB_bf(b, WGb + (long)(n0 + 16) * 256 + k0, 256); aG[1] = wmma_bf16(af, b, aG[1]);
  }
#pragma unroll
  for (int r = 0; r < 8; ++r) {
    int ml = r + 8 * half;
    float xv = x[(bt0 + ml) * 23 + f];
#pragma unroll
    for (int i = 0; i < 2; ++i) {
      int n = n0 + i * 16 + col;
      float hF = aF[i][r] + b2[f * 256 + n];
      float hG = aG[i][r] + bg[f * 256 + n];
      float hv = hF * sigmoidf_(hG);
      float xs = xv * wsk[f * 256 + n] + bsk[f * 256 + n];
      s[ml][n] = xs + hv;
    }
  }
  __syncthreads();
  float mx = fw[0];
  for (int i = 1; i < 23; ++i) mx = fmaxf(mx, fw[i]);
  float se = 0.f;
  for (int i = 0; i < 23; ++i) se += __expf(fw[i] - mx);
  float sw = __expf(fw[f] - mx) / se;
  for (int rr = 2 * wave; rr <= 2 * wave + 1; ++rr) {
    float sum = 0.f, sq = 0.f;
#pragma unroll
    for (int j = 0; j < 8; ++j) { float v = s[rr][lane + 32 * j]; sum += v; sq += v * v; }
#pragma unroll
    for (int m_ = 16; m_ >= 1; m_ >>= 1) {
      sum += __shfl_xor(sum, m_, 32);
      sq  += __shfl_xor(sq,  m_, 32);
    }
    float mean = sum * (1.f / 256.f);
    float var  = sq  * (1.f / 256.f) - mean * mean;
    float rstd = rsqrtf(var + 1e-5f);
    long bt = bt0 + rr;
#pragma unroll
    for (int j = 0; j < 8; ++j) {
      int n = lane + 32 * j;
      float v = (s[rr][n] - mean) * rstd * lng[f * 256 + n] + lnb[f * 256 + n];
      proc[bt * 5888 + f * 256 + n] = f2bf(v * sw);
    }
  }
}

// ---------------------------------------------------------------------------
// GRN tail: h = (A@W2+b2)*sigmoid(A@Wg+bg); out = LN(xs + h). N=K=256.
// W2/Wg: bf16 N-major. Grid: (M/16); block 256.
// ---------------------------------------------------------------------------
__global__ __launch_bounds__(256) void k_grn_tail(
    const unsigned short* A, const unsigned short* W2, const float* b2,
    const unsigned short* Wg, const float* bg, const float* xs,
    const float* lng, const float* lnb, float* outF, unsigned short* outBF) {
  __shared__ float s[16][256];
  const long m0 = (long)blockIdx.x * 16;
  int wave = threadIdx.x >> 5, lane = threadIdx.x & 31;
  int half = lane >> 4, col = lane & 15, n0 = wave * 32;
  v8f aF[2] = {vzero8(), vzero8()}, aG[2] = {vzero8(), vzero8()};
  const unsigned short* Ab = A + m0 * 256;
  for (int k0 = 0; k0 < 256; k0 += 32) {
    v16us af, b;
    ldA(af, Ab + k0, 256);
    ldB_bf(b, W2 + (long)(n0)      * 256 + k0, 256); aF[0] = wmma_bf16(af, b, aF[0]);
    ldB_bf(b, W2 + (long)(n0 + 16) * 256 + k0, 256); aF[1] = wmma_bf16(af, b, aF[1]);
    ldB_bf(b, Wg + (long)(n0)      * 256 + k0, 256); aG[0] = wmma_bf16(af, b, aG[0]);
    ldB_bf(b, Wg + (long)(n0 + 16) * 256 + k0, 256); aG[1] = wmma_bf16(af, b, aG[1]);
  }
#pragma unroll
  for (int r = 0; r < 8; ++r) {
    int ml = r + 8 * half;
#pragma unroll
    for (int i = 0; i < 2; ++i) {
      int n = n0 + i * 16 + col;
      float hv = (aF[i][r] + b2[n]) * sigmoidf_(aG[i][r] + bg[n]);
      s[ml][n] = xs[(m0 + ml) * 256 + n] + hv;
    }
  }
  __syncthreads();
  for (int rr = 2 * wave; rr <= 2 * wave + 1; ++rr) {
    float sum = 0.f, sq = 0.f;
#pragma unroll
    for (int j = 0; j < 8; ++j) { float v = s[rr][lane + 32 * j]; sum += v; sq += v * v; }
#pragma unroll
    for (int m_ = 16; m_ >= 1; m_ >>= 1) {
      sum += __shfl_xor(sum, m_, 32);
      sq  += __shfl_xor(sq,  m_, 32);
    }
    float mean = sum * (1.f / 256.f);
    float rstd = rsqrtf(sq * (1.f / 256.f) - mean * mean + 1e-5f);
    long m = m0 + rr;
#pragma unroll
    for (int j = 0; j < 8; ++j) {
      int n = lane + 32 * j;
      float v = (s[rr][n] - mean) * rstd * lng[n] + lnb[n];
      if (outF)  outF[m * 256 + n] = v;
      if (outBF) outBF[m * 256 + n] = f2bf(v);
    }
  }
}

// ---------------------------------------------------------------------------
// Persistent LSTM scan. One 1024-thread block per direction (grid.x = ndir).
// 64 batch rows, H=256, gates 1024. h in LDS (bf16, WMMA A source), c in regs,
// gate pre-activations staged in 256KB LDS. whh pre-converted bf16 (N-major).
// Dynamic LDS = 64*1024*4 + 64*264*2 = 295936 B (<= 320KB/WGP).
// ---------------------------------------------------------------------------
struct ScanArgs {
  const float* xg; const unsigned short* whh;
  const float* h0; const float* c0;
  unsigned short* ysbf; float* ysf;
  float* hT; float* cT;
  int reverse; int col0;
};
#define SCAN_LDS (64 * 1024 * 4 + 64 * 264 * 2)

__global__ __launch_bounds__(1024) void k_lstm_scan(ScanArgs A0, ScanArgs A1,
                                                    int T, int ldy) {
  ScanArgs a = (blockIdx.x == 0) ? A0 : A1;
  extern __shared__ char smem[];
  float* gbuf = (float*)smem;                                  // 64 x 1024
  unsigned short* hbuf = (unsigned short*)(smem + 64 * 1024 * 4);
  const int LDH = 264;
  int tid = threadIdx.x;
  int wave = tid >> 5, lane = tid & 31, half = lane >> 4, col = lane & 15;
  int n = tid & 255, mg = tid >> 8;
  float c[16];
#pragma unroll
  for (int rr = 0; rr < 16; ++rr) {
    int m = mg * 16 + rr;
    float hv = a.h0 ? a.h0[m * 256 + n] : 0.f;
    c[rr] = a.c0 ? a.c0[m * 256 + n] : 0.f;
    hbuf[m * LDH + n] = f2bf(hv);
  }
  __syncthreads();
  int nt0 = wave * 32;
  for (int t = 0; t < T; ++t) {
    int te = a.reverse ? (T - 1 - t) : t;
    v8f acc[4][2];
#pragma unroll
    for (int mt = 0; mt < 4; ++mt) { acc[mt][0] = vzero8(); acc[mt][1] = vzero8(); }
    for (int k0 = 0; k0 < 256; k0 += 32) {
      v16us b0, b1;
      ldB_bf(b0, a.whh + (long)nt0 * 256 + k0, 256);
      ldB_bf(b1, a.whh + (long)(nt0 + 16) * 256 + k0, 256);
#pragma unroll
      for (int mt = 0; mt < 4; ++mt) {
        v16us af;
        ldA(af, hbuf + (mt * 16) * LDH + k0, LDH);
        acc[mt][0] = wmma_bf16(af, b0, acc[mt][0]);
        acc[mt][1] = wmma_bf16(af, b1, acc[mt][1]);
      }
    }
#pragma unroll
    for (int mt = 0; mt < 4; ++mt)
#pragma unroll
      for (int i = 0; i < 2; ++i)
#pragma unroll
        for (int r = 0; r < 8; ++r)
          gbuf[(mt * 16 + r + 8 * half) * 1024 + nt0 + i * 16 + col] = acc[mt][i][r];
    __syncthreads();
#pragma unroll
    for (int rr = 0; rr < 16; ++rr) {
      int m = mg * 16 + rr;
      const float* xr = a.xg + ((long)m * T + te) * 1024;
      float gi = gbuf[m * 1024 + n]       + xr[n];
      float gf = gbuf[m * 1024 + 256 + n] + xr[256 + n];
      float gg = gbuf[m * 1024 + 512 + n] + xr[512 + n];
      float go = gbuf[m * 1024 + 768 + n] + xr[768 + n];
      float cn = sigmoidf_(gf) * c[rr] + sigmoidf_(gi) * tanhf(gg);
      float hn = sigmoidf_(go) * tanhf(cn);
      c[rr] = cn;
      long oi = ((long)m * T + te) * ldy + a.col0 + n;
      a.ysbf[oi] = f2bf(hn);
      if (a.ysf) a.ysf[oi] = hn;
      hbuf[m * LDH + n] = f2bf(hn);
    }
    __syncthreads();
  }
  if (a.hT) {
#pragma unroll
    for (int rr = 0; rr < 16; ++rr) {
      int m = mg * 16 + rr;
      a.hT[m * 256 + n] = bf2f(hbuf[m * LDH + n]);
      a.cT[m * 256 + n] = c[rr];
    }
  }
}

// ---------------------------------------------------------------------------
// Small helpers
// ---------------------------------------------------------------------------
__global__ __launch_bounds__(256) void k_avg(const float* a, const float* b,
                                             float* o, int nE) {
  int i = blockIdx.x * 256 + threadIdx.x;
  if (i < nE) o[i] = 0.5f * (a[i] + b[i]);
}

__global__ __launch_bounds__(256) void k_tile_last(const float* encF,
                                                   unsigned short* dec0) {
  int i = blockIdx.x * 256 + threadIdx.x;   // < 1536*256
  int nn = i & 255;
  int row = i >> 8;                         // b*24 + t
  int b = row / 24;
  dec0[i] = f2bf(encF[((long)b * 168 + 167) * 256 + nn]);
}

// Attention core (tiny): grid (Tq, NH, B), block 64.
__global__ __launch_bounds__(64) void k_attn(const float* q, const float* kk,
                                             const float* vv, float* oF,
                                             unsigned short* oBF, int Tq, int Tk) {
  __shared__ float sc[192];
  __shared__ float red[64];
  __shared__ float smx, ssum;
  int b = blockIdx.z, nh = blockIdx.y, qi = blockIdx.x, tid = threadIdx.x;
  const float* qr = q + ((long)b * Tq + qi) * 256 + nh * 32;
  float lmax = -1e30f;
  for (int kp = tid; kp < Tk; kp += 64) {
    const float* kr = kk + ((long)b * Tk + kp) * 256 + nh * 32;
    float s = 0.f;
#pragma unroll
    for (int d = 0; d < 32; ++d) s += qr[d] * kr[d];
    s *= 0.17677669529663687f;   // 1/sqrt(32)
    sc[kp] = s;
    lmax = fmaxf(lmax, s);
  }
  red[tid] = lmax; __syncthreads();
  if (tid == 0) { float m = red[0]; for (int i = 1; i < 64; ++i) m = fmaxf(m, red[i]); smx = m; }
  __syncthreads();
  float lsum = 0.f;
  for (int kp = tid; kp < Tk; kp += 64) { float e = __expf(sc[kp] - smx); sc[kp] = e; lsum += e; }
  red[tid] = lsum; __syncthreads();
  if (tid == 0) { float s2 = 0.f; for (int i = 0; i < 64; ++i) s2 += red[i]; ssum = s2; }
  __syncthreads();
  if (tid < 32) {
    float inv = 1.f / ssum, acc = 0.f;
    for (int kp = 0; kp < Tk; ++kp)
      acc += sc[kp] * vv[((long)b * Tk + kp) * 256 + nh * 32 + tid];
    acc *= inv;
    long oi = ((long)b * Tq + qi) * 256 + nh * 32 + tid;
    oF[oi] = acc; oBF[oi] = f2bf(acc);
  }
}

// Final: out[b,t,o,q] = sum_h dec[bt,h]*out_w[o,h,q] + out_b[o,q]
__global__ __launch_bounds__(256) void k_out_proj(const float* dec,
                                                  const float* w,
                                                  const float* bia, float* out) {
  int i = blockIdx.x * 256 + threadIdx.x;
  if (i >= 64 * 24 * 13 * 3) return;
  int qq = i % 3, o = (i / 3) % 13;
  long row = i / 39;
  float acc = bia[o * 3 + qq];
  const float* dr = dec + row * 256;
  const float* wr = w + (long)o * 256 * 3 + qq;
  for (int h = 0; h < 256; ++h) acc += dr[h] * wr[h * 3];
  out[i] = acc;
}

// ---------------------------------------------------------------------------
// Host orchestration
// ---------------------------------------------------------------------------
extern "C" void kernel_launch(void* const* d_in, const int* in_sizes, int n_in,
                              void* d_out, int out_size, void* d_ws,
                              size_t ws_size, hipStream_t stream) {
  (void)in_sizes; (void)n_in; (void)out_size; (void)ws_size;
  const float* x = (const float*)d_in[0];
#define P(i) ((const float*)d_in[i])
  const long BT = 64L * 168;      // 10752
  const long BFS = 64L * 24;      // 1536
  const size_t MB = 1024 * 1024;
  char* W = (char*)d_ws;

  // Region 0 (0..132MB): h1 bf16 during VSN; reused for xg / ca-kv after.
  unsigned short* h1bf = (unsigned short*)(W);
  float* xgf = (float*)(W);                 // 44MB
  float* xgb = (float*)(W + 48 * MB);       // 44MB
  float* xgd = (float*)(W);                 // 6.3MB (decoder)
  float* kca = (float*)(W + 8 * MB);        // 11MB
  float* vca = (float*)(W + 20 * MB);       // 11MB
  // Region 1 (132..264MB): proc bf16 during VSN/gc; reused downstream.
  unsigned short* procbf = (unsigned short*)(W + 132 * MB);
  unsigned short* inps[3] = {(unsigned short*)(W + 132 * MB),
                             (unsigned short*)(W + 144 * MB),
                             (unsigned short*)(W + 156 * MB)};
  float* encF  = (float*)(W + 168 * MB);
  unsigned short* encBF = (unsigned short*)(W + 180 * MB);
  unsigned short* decbf[2] = {(unsigned short*)(W + 186 * MB),
                              (unsigned short*)(W + 187 * MB)};
  float* decf[2] = {(float*)(W + 188 * MB), (float*)(W + 190 * MB)};
  float* saq = (float*)(W + 192 * MB);
  float* sak = (float*)(W + 194 * MB);
  float* sav = (float*)(W + 196 * MB);
  float* aoF = (float*)(W + 198 * MB);
  unsigned short* aoBF = (unsigned short*)(W + 200 * MB);
  float* dec2F = (float*)(W + 202 * MB);
  unsigned short* dec2BF = (unsigned short*)(W + 204 * MB);
  float* dec3F = (float*)(W + 206 * MB);
  unsigned short* dec3BF = (unsigned short*)(W + 208 * MB);
  float* caq = (float*)(W + 210 * MB);
  unsigned short* h1gp = (unsigned short*)(W + 212 * MB);
  float* gpF = (float*)(W + 214 * MB);
  // Region 2 (264MB..): gc intermediates + LSTM states.
  unsigned short* h1g = (unsigned short*)(W + 264 * MB);
  float* xsF = (float*)(W + 270 * MB);
  unsigned short* selBF = (unsigned short*)(W + 282 * MB);
  float* st = (float*)(W + 288 * MB);
  float *hf = st, *cf = st + 16384, *hb = st + 2 * 16384, *cb = st + 3 * 16384;
  float* havg[3] = {st + 4 * 16384, st + 5 * 16384, st + 6 * 16384};
  float* cavg[3] = {st + 7 * 16384, st + 8 * 16384, st + 9 * 16384};

  // Region 3 (296MB..): bf16 weight cache (~26MB), bump-allocated.
  size_t woff = 296 * MB;
  auto WB = [&](size_t elems) {
    unsigned short* p = (unsigned short*)(W + woff);
    woff += ((elems * 2 + 255) & ~(size_t)255);
    return p;
  };
  auto convCopy = [&](int idx, size_t elems) {
    unsigned short* p = WB(elems);
    k_conv_copy<<<dim3((unsigned)((elems + 255) / 256)), 256, 0, stream>>>(
        P(idx), p, (int)elems);
    return p;
  };
  auto convTr = [&](int idx, int K, int N, int batch) {
    unsigned short* p = WB((size_t)K * N * batch);
    k_conv_tr<<<dim3((unsigned)(((long)K * N + 255) / 256), batch), 256, 0,
                stream>>>(P(idx), p, K, N);
    return p;
  };

  // ---- 0. Weight pre-conversion to bf16 (N-major, K contiguous) ----
  unsigned short* wb_vsn_fc2  = convTr(3, 256, 256, 23);
  unsigned short* wb_vsn_gate = convTr(5, 256, 256, 23);
  unsigned short* wb_gc_fc1   = convTr(12, 5888, 256, 1);
  unsigned short* wb_gc_skip  = convTr(28, 5888, 256, 1);
  unsigned short* wb_gc_fc2   = convTr(14, 256, 256, 1);
  unsigned short* wb_gc_gate  = convTr(16, 256, 256, 1);
  unsigned short* wb_gp_fc1   = convTr(20, 256, 256, 1);
  unsigned short* wb_gp_fc2   = convTr(22, 256, 256, 1);
  unsigned short* wb_gp_gate  = convTr(24, 256, 256, 1);
  unsigned short* wb_proj     = convTr(66, 512, 256, 1);
  unsigned short* wb_att[8];
  const int attIdx[8] = {68, 69, 70, 71, 76, 77, 78, 79};
  for (int i = 0; i < 8; ++i) wb_att[i] = convTr(attIdx[i], 256, 256, 1);
  // wih / whh are already (N,K) row-major -> straight bf16 copy.
  unsigned short *wb_ewih[3][2], *wb_ewhh[3][2], *wb_dwih[3], *wb_dwhh[3];
  const int wihIdx[3] = {30, 42, 54};
  const int dIdx[3] = {38, 50, 62};
  const size_t wihElems[3] = {1024 * 256, 1024 * 512, 1024 * 512};
  for (int l = 0; l < 3; ++l) {
    wb_ewih[l][0] = convCopy(wihIdx[l],     wihElems[l]);
    wb_ewhh[l][0] = convCopy(wihIdx[l] + 1, 1024 * 256);
    wb_ewih[l][1] = convCopy(wihIdx[l] + 4, wihElems[l]);
    wb_ewhh[l][1] = convCopy(wihIdx[l] + 5, 1024 * 256);
    wb_dwih[l]    = convCopy(dIdx[l],       1024 * 256);
    wb_dwhh[l]    = convCopy(dIdx[l] + 1,   1024 * 256);
  }

  // ---- 1. VSN (fused dual-GEMM + gate + skip + LN + feature softmax) ----
  k_vsn_h1<<<dim3((unsigned)(BT * 23)), 256, 0, stream>>>(x, P(1), P(2), h1bf);
  k_vsn_fused<<<dim3((unsigned)(BT / 16), 23), 256, 0, stream>>>(
      h1bf, x, wb_vsn_fc2, P(4), wb_vsn_gate, P(6), P(7), P(8), P(9), P(10),
      P(11), procbf);

  // ---- 2. gc GRN (K = 5888) ----
  k_gemm<<<dim3((unsigned)(BT / 32), 4), 256, 0, stream>>>(
      procbf, 5888, wb_gc_fc1, P(13), nullptr, nullptr, 0, 1,
      nullptr, h1g, 256, 5888);
  k_gemm<<<dim3((unsigned)(BT / 32), 4), 256, 0, stream>>>(
      procbf, 5888, wb_gc_skip, P(29), nullptr, nullptr, 0, 0,
      xsF, nullptr, 256, 5888);
  k_grn_tail<<<dim3((unsigned)(BT / 16)), 256, 0, stream>>>(
      h1g, wb_gc_fc2, P(15), wb_gc_gate, P(17), xsF, P(18), P(19),
      nullptr, selBF);

  // ---- 3. Bidirectional LSTM encoder (3 layers) ----
  const unsigned short* inp = selBF;
  int din = 256;
  for (int l = 0; l < 3; ++l) {
    int fi = wihIdx[l], bi = fi + 4;
    k_gemm<<<dim3((unsigned)(BT / 32), 16), 256, 0, stream>>>(
        inp, din, wb_ewih[l][0], P(fi + 2), P(fi + 3), nullptr, 0, 0,
        xgf, nullptr, 1024, din);
    k_gemm<<<dim3((unsigned)(BT / 32), 16), 256, 0, stream>>>(
        inp, din, wb_ewih[l][1], P(bi + 2), P(bi + 3), nullptr, 0, 0,
        xgb, nullptr, 1024, din);
    ScanArgs s0 = {xgf, wb_ewhh[l][0], nullptr, nullptr, inps[l], nullptr,
                   hf, cf, 0, 0};
    ScanArgs s1 = {xgb, wb_ewhh[l][1], nullptr, nullptr, inps[l], nullptr,
                   hb, cb, 1, 256};
    k_lstm_scan<<<dim3(2), 1024, SCAN_LDS, stream>>>(s0, s1, 168, 512);
    k_avg<<<dim3(64), 256, 0, stream>>>(hf, hb, havg[l], 16384);
    k_avg<<<dim3(64), 256, 0, stream>>>(cf, cb, cavg[l], 16384);
    inp = inps[l];
    din = 512;
  }
  // proj to enc_out
  k_gemm<<<dim3((unsigned)(BT / 32), 4), 256, 0, stream>>>(
      inps[2], 512, wb_proj, P(67), nullptr, nullptr, 0, 0,
      encF, encBF, 256, 512);

  // ---- 4. Decoder LSTM (3 layers, init from averaged enc states) ----
  k_tile_last<<<dim3((unsigned)(BFS)), 256, 0, stream>>>(encF, decbf[0]);
  const unsigned short* dp = decbf[0];
  float* decFfin = nullptr;
  const unsigned short* decBFfin = nullptr;
  for (int l = 0; l < 3; ++l) {
    int di = dIdx[l];
    k_gemm<<<dim3((unsigned)(BFS / 32), 16), 256, 0, stream>>>(
        dp, 256, wb_dwih[l], P(di + 2), P(di + 3), nullptr, 0, 0,
        xgd, nullptr, 1024, 256);
    unsigned short* ob = decbf[(l + 1) & 1];
    float* of = decf[(l + 1) & 1];
    ScanArgs sd = {xgd, wb_dwhh[l], havg[l], cavg[l], ob, of,
                   nullptr, nullptr, 0, 0};
    k_lstm_scan<<<dim3(1), 1024, SCAN_LDS, stream>>>(sd, sd, 24, 256);
    dp = ob; decFfin = of; decBFfin = ob;
  }

  // ---- 5. Self-attention + residual ----
  k_gemm<<<dim3((unsigned)(BFS / 32), 4), 256, 0, stream>>>(
      decBFfin, 256, wb_att[0], P(72), nullptr, nullptr, 0, 0, saq, nullptr, 256, 256);
  k_gemm<<<dim3((unsigned)(BFS / 32), 4), 256, 0, stream>>>(
      decBFfin, 256, wb_att[1], P(73), nullptr, nullptr, 0, 0, sak, nullptr, 256, 256);
  k_gemm<<<dim3((unsigned)(BFS / 32), 4), 256, 0, stream>>>(
      decBFfin, 256, wb_att[2], P(74), nullptr, nullptr, 0, 0, sav, nullptr, 256, 256);
  k_attn<<<dim3(24, 8, 64), 64, 0, stream>>>(saq, sak, sav, aoF, aoBF, 24, 24);
  k_gemm<<<dim3((unsigned)(BFS / 32), 4), 256, 0, stream>>>(
      aoBF, 256, wb_att[3], P(75), nullptr, decFfin, 256, 0,
      dec2F, dec2BF, 256, 256);

  // ---- 6. Cross-attention + residual ----
  k_gemm<<<dim3((unsigned)(BFS / 32), 4), 256, 0, stream>>>(
      dec2BF, 256, wb_att[4], P(80), nullptr, nullptr, 0, 0, caq, nullptr, 256, 256);
  k_gemm<<<dim3((unsigned)(BT / 32), 4), 256, 0, stream>>>(
      encBF, 256, wb_att[5], P(81), nullptr, nullptr, 0, 0, kca, nullptr, 256, 256);
  k_gemm<<<dim3((unsigned)(BT / 32), 4), 256, 0, stream>>>(
      encBF, 256, wb_att[6], P(82), nullptr, nullptr, 0, 0, vca, nullptr, 256, 256);
  k_attn<<<dim3(24, 8, 64), 64, 0, stream>>>(caq, kca, vca, aoF, aoBF, 24, 168);
  k_gemm<<<dim3((unsigned)(BFS / 32), 4), 256, 0, stream>>>(
      aoBF, 256, wb_att[7], P(83), nullptr, dec2F, 256, 0,
      dec3F, dec3BF, 256, 256);

  // ---- 7. gp GRN (skip=False) + output projection ----
  k_gemm<<<dim3((unsigned)(BFS / 32), 4), 256, 0, stream>>>(
      dec3BF, 256, wb_gp_fc1, P(21), nullptr, nullptr, 0, 1,
      nullptr, h1gp, 256, 256);
  k_grn_tail<<<dim3((unsigned)(BFS / 16)), 256, 0, stream>>>(
      h1gp, wb_gp_fc2, P(23), wb_gp_gate, P(25), dec3F, P(26), P(27),
      gpF, nullptr);
  k_out_proj<<<dim3((64 * 24 * 13 * 3 + 255) / 256), 256, 0, stream>>>(
      gpF, P(84), P(85), (float*)d_out);
#undef P
}
